// MENL_module_40389872451627
// MI455X (gfx1250) — matrix operand order repaired and checked by reading, not compiled, not run on hardware
//
#include <hip/hip_runtime.h>
#include <hip/hip_bf16.h>
#include <stdint.h>

// B=16, C=512, N=4096, R=2 -> CR=256.  All GEMM dims divide the 128x128x32 tile.
typedef __bf16 bf16;
typedef __attribute__((ext_vector_type(16))) __bf16 v16bf;
typedef __attribute__((ext_vector_type(8)))  __bf16 v8bf;
typedef __attribute__((ext_vector_type(4)))  __bf16 v4bf;
typedef __attribute__((ext_vector_type(8)))  float  v8f;

#define BN_EPS 1e-5f
#define LDSTRIDE 40            // bf16 elems: 80B rows -> 16B aligned, 20-bank spacing
#define TILE_ELEMS (128 * LDSTRIDE)

// ---------------------------------------------------------------- convert ---
__global__ __launch_bounds__(256)
void cvt_f32_to_bf16(const float4* __restrict__ src, v4bf* __restrict__ dst, long n4) {
    long i = (long)blockIdx.x * blockDim.x + threadIdx.x;
    long stride = (long)gridDim.x * blockDim.x;
    for (; i < n4; i += stride) {
        float4 f = src[i];
        v4bf o;
        o[0] = (bf16)f.x; o[1] = (bf16)f.y; o[2] = (bf16)f.z; o[3] = (bf16)f.w;
        dst[i] = o;
    }
}

// ------------------------------------------------------------ WMMA GEMM -----
// D[b] = A[b?] (MxK row-major bf16) * B[b] (KxN row-major; or NxK row-major
// used transposed when TRANSB).  Epilogues:
//   EPI 0: out_bf16 = acc + bias[m]                       (Q/K/V projections)
//   EPI 1: out_f32  = acc                                 (corr)
//   EPI 2: out_bf16 = acc                                 (attn @ V)
//   EPI 3: out_f32  = (acc+bias)*scale[m]+shift[m]+res    (final proj + BN + x)
// Block tile 128(M) x 128(N), BK=32; 8 waves, each wave 32x64 via 2x4
// v_wmma_f32_16x16x32_bf16 accumulators.  B tile kept [n][k] in LDS so all
// fragment reads are contiguous ds_load_b128.  Double-buffered LDS with a
// single barrier per K-step; next global tile prefetched into registers while
// the current tile is consumed.
template <int EPI, bool TRANSB>
__global__ __launch_bounds__(256)
void gemm_bf16_wmma(const bf16* __restrict__ A, long strideA,
                    const bf16* __restrict__ Bmat, long strideB,
                    const float* __restrict__ bias,
                    bf16* __restrict__ outB, float* __restrict__ outF, long strideO,
                    const float* __restrict__ gamma, const float* __restrict__ beta,
                    const float* __restrict__ rmean, const float* __restrict__ rvar,
                    const float* __restrict__ resid,
                    int M, int Nd, int Kd) {
    __shared__ __align__(16) bf16 As[2 * TILE_ELEMS];
    __shared__ __align__(16) bf16 Bs[2 * TILE_ELEMS];

    const int b = blockIdx.z;
    const bf16* Ab = A + (long)b * strideA;
    const bf16* Bb = Bmat + (long)b * strideB;

    const int t = threadIdx.x;
    const int wave = t >> 5, lane = t & 31;
    const int wm = (wave >> 1) * 32;           // 4 wave-rows   (M)
    const int wn = (wave & 1) * 64;            // 2 wave-cols   (N)
    const int row0 = blockIdx.y * 128, col0 = blockIdx.x * 128;

    // A tile 128(m) x 32(k): thread -> row t>>1, 16-elem half (t&1)*16
    const int ar = t >> 1, ac = (t & 1) * 16;
    // TRANSB B tile 128(n) x 32(k): same mapping on rows of Bb ([Nd,Kd])
    // !TRANSB B tile 32(k) x 128(n): lane = k-row, wave picks 16-n block;
    //   scatter-store to Bs[n][k] is conflict-free (consecutive lanes ->
    //   consecutive k inside one 64B row segment).
    const int nk_k = lane, nk_n = wave * 16;

    v8f acc[2][4] = {};

    v8bf a0, a1, b0, b1;
    auto loadTiles = [&](int k0) {
        const v8bf* ga = reinterpret_cast<const v8bf*>(Ab + (long)(row0 + ar) * Kd + k0 + ac);
        a0 = ga[0]; a1 = ga[1];
        if (TRANSB) {
            const v8bf* gb = reinterpret_cast<const v8bf*>(Bb + (long)(col0 + ar) * Kd + k0 + ac);
            b0 = gb[0]; b1 = gb[1];
        } else {
            const v8bf* gb = reinterpret_cast<const v8bf*>(Bb + (long)(k0 + nk_k) * Nd + col0 + nk_n);
            b0 = gb[0]; b1 = gb[1];
        }
    };
    auto storeTiles = [&](int pb) {
        bf16* AsP = As + pb * TILE_ELEMS;
        bf16* BsP = Bs + pb * TILE_ELEMS;
        v8bf* sA = reinterpret_cast<v8bf*>(&AsP[ar * LDSTRIDE + ac]);
        sA[0] = a0; sA[1] = a1;
        if (TRANSB) {
            v8bf* sB = reinterpret_cast<v8bf*>(&BsP[ar * LDSTRIDE + ac]);
            sB[0] = b0; sB[1] = b1;
        } else {
#pragma unroll
            for (int i = 0; i < 8; ++i) {
                BsP[(nk_n + i) * LDSTRIDE + nk_k] = b0[i];
                BsP[(nk_n + 8 + i) * LDSTRIDE + nk_k] = b1[i];
            }
        }
    };
    // ISA 7.12.2 fragment coordinates
    const int am = lane & 15, akb = (lane >> 4) * 8;    // A: row, K base 0/8 (+16)
    const int bn = lane & 15, bkb = (lane >> 4) * 8;    // B: col, K base 0/8 (+16)
    auto compute = [&](int pb) {
        const bf16* AsP = As + pb * TILE_ELEMS;
        const bf16* BsP = Bs + pb * TILE_ELEMS;
        v16bf af[2], bf[4];
#pragma unroll
        for (int ti = 0; ti < 2; ++ti) {
            const bf16* p = &AsP[(wm + ti * 16 + am) * LDSTRIDE];
            v8bf lo = *reinterpret_cast<const v8bf*>(p + akb);
            v8bf hi = *reinterpret_cast<const v8bf*>(p + akb + 16);
            af[ti] = __builtin_shufflevector(lo, hi, 0, 1, 2, 3, 4, 5, 6, 7,
                                             8, 9, 10, 11, 12, 13, 14, 15);
        }
#pragma unroll
        for (int tj = 0; tj < 4; ++tj) {
            const bf16* p = &BsP[(wn + tj * 16 + bn) * LDSTRIDE];
            v8bf lo = *reinterpret_cast<const v8bf*>(p + bkb);
            v8bf hi = *reinterpret_cast<const v8bf*>(p + bkb + 16);
            bf[tj] = __builtin_shufflevector(lo, hi, 0, 1, 2, 3, 4, 5, 6, 7,
                                             8, 9, 10, 11, 12, 13, 14, 15);
        }
#pragma unroll
        for (int ti = 0; ti < 2; ++ti)
#pragma unroll
            for (int tj = 0; tj < 4; ++tj)
                acc[ti][tj] = __builtin_amdgcn_wmma_f32_16x16x32_bf16(
                    false, af[ti], false, bf[tj], (short)0, acc[ti][tj], false, false);
    };

    loadTiles(0);
    storeTiles(0);
    __syncthreads();
    int cur = 0;
    for (int k0 = 32; k0 < Kd; k0 += 32) {
        loadTiles(k0);          // prefetch next tile into regs
        compute(cur);           // WMMA from current buffer (hides load latency)
        storeTiles(cur ^ 1);    // park prefetched tile in the inactive buffer
        __syncthreads();        // single barrier per K-step
        cur ^= 1;
    }
    compute(cur);

    // Epilogue. C/D layout: vgpr r, lane -> m = r + 8*(lane>>4), n = lane&15
    const long outBase = (long)b * strideO;
#pragma unroll
    for (int ti = 0; ti < 2; ++ti) {
#pragma unroll
        for (int r = 0; r < 8; ++r) {
            int m = row0 + wm + ti * 16 + r + 8 * (lane >> 4);
            float bi = (EPI == 0 || EPI == 3) ? bias[m] : 0.f;
            float sc = 1.f, sh = 0.f;
            if (EPI == 3) {
                sc = gamma[m] * rsqrtf(rvar[m] + BN_EPS);
                sh = beta[m] - rmean[m] * sc;
            }
#pragma unroll
            for (int tj = 0; tj < 4; ++tj) {
                int n = col0 + wn + tj * 16 + (lane & 15);
                long idx = outBase + (long)m * Nd + n;
                float v = acc[ti][tj][r] + bi;
                if (EPI == 0 || EPI == 2)      outB[idx] = (bf16)v;
                else if (EPI == 1)             outF[idx] = v;
                else                           outF[idx] = v * sc + sh + resid[idx];
            }
        }
    }
}

// ------------------------------------------------------------- softmax ------
// One wave32 per row of 512; f32 in, bf16 out.
__global__ __launch_bounds__(256)
void softmax_rows(const float* __restrict__ src, bf16* __restrict__ dst,
                  int rows, int cols) {
    int row = blockIdx.x * (blockDim.x >> 5) + (threadIdx.x >> 5);
    if (row >= rows) return;
    int lane = threadIdx.x & 31;
    const float* p = src + (long)row * cols;
    float mx = -3.4e38f;
    for (int c = lane; c < cols; c += 32) mx = fmaxf(mx, p[c]);
#pragma unroll
    for (int off = 16; off > 0; off >>= 1) mx = fmaxf(mx, __shfl_xor(mx, off, 32));
    float sum = 0.f;
    for (int c = lane; c < cols; c += 32) sum += __expf(p[c] - mx);
#pragma unroll
    for (int off = 16; off > 0; off >>= 1) sum += __shfl_xor(sum, off, 32);
    float inv = 1.f / sum;
    bf16* q = dst + (long)row * cols;
    for (int c = lane; c < cols; c += 32) q[c] = (bf16)(__expf(p[c] - mx) * inv);
}

// -------------------------------------------------------------- launch ------
extern "C" void kernel_launch(void* const* d_in, const int* in_sizes, int n_in,
                              void* d_out, int out_size, void* d_ws, size_t ws_size,
                              hipStream_t stream) {
    constexpr int Bb = 16, C = 512, N = 4096, CR = 256, C2 = 1024;

    const float* x     = (const float*)d_in[0];
    const float* xf    = (const float*)d_in[1];
    const float* Wq    = (const float*)d_in[2];
    const float* bq    = (const float*)d_in[3];
    const float* Wk    = (const float*)d_in[4];
    const float* bk    = (const float*)d_in[5];
    const float* Wv    = (const float*)d_in[6];
    const float* bv    = (const float*)d_in[7];
    const float* Wr    = (const float*)d_in[8];
    const float* br    = (const float*)d_in[9];
    const float* gamma = (const float*)d_in[10];
    const float* beta  = (const float*)d_in[11];
    const float* rmean = (const float*)d_in[12];
    const float* rvar  = (const float*)d_in[13];
    float* out = (float*)d_out;

    char* ws = (char*)d_ws;
    auto take = [&](long bytes) {
        char* p = ws;
        ws += (bytes + 255) & ~255L;
        return p;
    };
    bf16* xb    = (bf16*)take((long)Bb * C  * N * 2);
    bf16* xfb   = (bf16*)take((long)Bb * C2 * N * 2);
    bf16* Wqb   = (bf16*)take((long)CR * C  * 2);
    bf16* Wkb   = (bf16*)take((long)C  * C2 * 2);
    bf16* Wvb   = (bf16*)take((long)C  * C2 * 2);
    bf16* Wrb   = (bf16*)take((long)C  * CR * 2);
    bf16* Qb    = (bf16*)take((long)Bb * CR * N * 2);
    bf16* Kb    = (bf16*)take((long)Bb * C  * N * 2);
    bf16* Vb    = (bf16*)take((long)Bb * C  * N * 2);
    float* corr = (float*)take((long)Bb * CR * C * 4);
    bf16* attn  = (bf16*)take((long)Bb * CR * C * 2);
    bf16* yb    = (bf16*)take((long)Bb * CR * N * 2);

    dim3 blk(256);
    auto cvt = [&](const float* s, bf16* d, long n) {
        long n4 = n >> 2;
        int g = (int)((n4 + 256 * 4 - 1) / (256 * 4));
        if (g > 4096) g = 4096;
        cvt_f32_to_bf16<<<g, blk, 0, stream>>>((const float4*)s, (v4bf*)d, n4);
    };
    cvt(x,  xb,  (long)Bb * C * N);
    cvt(xf, xfb, (long)Bb * C2 * N);
    cvt(Wq, Wqb, (long)CR * C);
    cvt(Wk, Wkb, (long)C * C2);
    cvt(Wv, Wvb, (long)C * C2);
    cvt(Wr, Wrb, (long)C * CR);

    // Q = Wq*x + bq  -> bf16 [B, 256, 4096]
    gemm_bf16_wmma<0, false><<<dim3(N / 128, CR / 128, Bb), blk, 0, stream>>>(
        Wqb, 0, xb, (long)C * N, bq, Qb, nullptr, (long)CR * N,
        nullptr, nullptr, nullptr, nullptr, nullptr, CR, N, C);
    // K = Wk*xf + bk -> bf16 [B, 512, 4096]
    gemm_bf16_wmma<0, false><<<dim3(N / 128, C / 128, Bb), blk, 0, stream>>>(
        Wkb, 0, xfb, (long)C2 * N, bk, Kb, nullptr, (long)C * N,
        nullptr, nullptr, nullptr, nullptr, nullptr, C, N, C2);
    // V = Wv*xf + bv -> bf16 [B, 512, 4096]
    gemm_bf16_wmma<0, false><<<dim3(N / 128, C / 128, Bb), blk, 0, stream>>>(
        Wvb, 0, xfb, (long)C2 * N, bv, Vb, nullptr, (long)C * N,
        nullptr, nullptr, nullptr, nullptr, nullptr, C, N, C2);
    // corr = Q @ K^T over n -> f32 [B, 256, 512]   (B operand = K, transposed)
    gemm_bf16_wmma<1, true><<<dim3(C / 128, CR / 128, Bb), blk, 0, stream>>>(
        Qb, (long)CR * N, Kb, (long)C * N, nullptr, nullptr, corr, (long)CR * C,
        nullptr, nullptr, nullptr, nullptr, nullptr, CR, C, N);
    // softmax over channels -> bf16 attn
    softmax_rows<<<dim3((Bb * CR) / 8), blk, 0, stream>>>(corr, attn, Bb * CR, C);
    // y = attn @ V -> bf16 [B, 256, 4096]
    gemm_bf16_wmma<2, false><<<dim3(N / 128, CR / 128, Bb), blk, 0, stream>>>(
        attn, (long)CR * C, Vb, (long)C * N, nullptr, yb, nullptr, (long)CR * N,
        nullptr, nullptr, nullptr, nullptr, nullptr, CR, N, C);
    // out = BN(Wr*y + br) + x -> f32 [B, 512, 4096]
    gemm_bf16_wmma<3, false><<<dim3(N / 128, C / 128, Bb), blk, 0, stream>>>(
        Wrb, 0, yb, (long)CR * N, br, nullptr, out, (long)C * N,
        gamma, beta, rmean, rvar, x, C, N, CR);

    (void)in_sizes; (void)n_in; (void)out_size; (void)ws_size;
}
